// MicroTransformer_80384607912317
// MI455X (gfx1250) — matrix operand order repaired and checked by reading, not compiled, not run on hardware
//
#include <hip/hip_runtime.h>

// ---------------------------------------------------------------------------
// MicroTransformer forward for MI455X (gfx1250, wave32, WMMA).
// All GEMMs: C[M,N] = A_bf16[M,K] @ Bt_bf16[N,K]^T, fp32 accumulation via
// V_WMMA_F32_16X16X32_BF16. Double-buffered LDS with async global->LDS loads
// (ASYNCcnt) via GLOBAL_LOAD_ASYNC_TO_LDS_B128.
// ---------------------------------------------------------------------------

typedef unsigned short u16;
typedef __bf16 bf16_t;
typedef bf16_t v16bf __attribute__((ext_vector_type(16)));
typedef float  v8f   __attribute__((ext_vector_type(8)));
typedef int    v4i   __attribute__((ext_vector_type(4)));

#define AS1 __attribute__((address_space(1)))
#define AS3 __attribute__((address_space(3)))

#if __has_builtin(__builtin_amdgcn_global_load_async_to_lds_b128) && \
    __has_builtin(__builtin_amdgcn_s_wait_asynccnt)
#define USE_ASYNC_LDS 1
#else
#define USE_ASYNC_LDS 0
#endif

#define D_MODEL 768
#define SEQ_T   2048
#define BATCH   2
#define NHEAD   12
#define DHEAD   64
#define D_FF    3072
#define NLAYER  4
#define VOCAB   32000
#define MROWS   (BATCH * SEQ_T)   // 4096

__device__ __forceinline__ u16 f32_to_bf16(float f) {
    unsigned int u = __float_as_uint(f);
    unsigned int lsb = (u >> 16) & 1u;
    u += 0x7fffu + lsb;                 // round to nearest even
    return (u16)(u >> 16);
}

// ---------------------------------------------------------------------------
// Generic WMMA GEMM, templated on wave geometry.
//   256 threads = 8 waves arranged WGM x WGN; each wave computes a
//   (TM*16) x (TN*16) tile; workgroup tile BM x BN, BK = 32.
// Variants used:  <2,4,4,2> -> 128x128 (8 WMMA/wave/K-step)
//                 <4,2,2,2> -> 128x64  (N=64 case: P@V)
// Bt is B stored transposed: Bt[n][k], row stride ldb.
// Epilogue: optional bias[N], ReLU, fp32 residual; fp32 or bf16 output.
// ---------------------------------------------------------------------------
#define GBK  32
#define LSTR 40   // LDS row stride (elements); 80B = multiple of 16B

union FragCast { v16bf b; uint4 q[2]; };

template <int WGM, int WGN, int TM, int TN>
__global__ __launch_bounds__(256) void k_gemm(
    const u16* __restrict__ A,  int lda,
    const u16* __restrict__ Bt, int ldb,
    float* outF, u16* outB, int ldc,
    const float* __restrict__ bias,
    const float* __restrict__ resid, int ldr,
    int K, int relu)
{
    constexpr int BM  = WGM * TM * 16;
    constexpr int BN  = WGN * TN * 16;
    constexpr int NA  = BM / 64;        // 16B chunks per thread for A tile
    constexpr int NB  = BN / 64;        // 16B chunks per thread for B tile

    __shared__ __align__(16) u16 As[2][BM * LSTR];
    __shared__ __align__(16) u16 Bs[2][BN * LSTR];

    const int tid  = threadIdx.x;
    const int lane = tid & 31;
    const int w    = tid >> 5;          // 0..7
    const int wm   = w / WGN;
    const int wn   = w % WGN;
    const int rowBase = blockIdx.y * BM;
    const int colBase = blockIdx.x * BN;

    const int lr16 = lane & 15;
    const bool hiHalf = lane >= 16;

    v8f acc[TM][TN] = {};

    const int nt = K / GBK;

    // ---- tile staging (global -> LDS), chunk = 8 bf16 = 16B per thread ----
    auto stage = [&](int it, int buf) {
        const int kt = it * GBK;
#if USE_ASYNC_LDS
        #pragma unroll
        for (int j = 0; j < NA; ++j) {
            const int idx = tid + j * 256;
            const int r = idx >> 2, c = (idx & 3) * 8;
            __builtin_amdgcn_global_load_async_to_lds_b128(
                (AS1 v4i*)(A + (size_t)(rowBase + r) * lda + kt + c),
                (AS3 v4i*)(&As[buf][r * LSTR + c]), 0, 0);
        }
        #pragma unroll
        for (int j = 0; j < NB; ++j) {
            const int idx = tid + j * 256;
            const int r = idx >> 2, c = (idx & 3) * 8;
            __builtin_amdgcn_global_load_async_to_lds_b128(
                (AS1 v4i*)(Bt + (size_t)(colBase + r) * ldb + kt + c),
                (AS3 v4i*)(&Bs[buf][r * LSTR + c]), 0, 0);
        }
#else
        #pragma unroll
        for (int j = 0; j < NA; ++j) {
            const int idx = tid + j * 256;
            const int r = idx >> 2, c = (idx & 3) * 8;
            uint4 x = *(const uint4*)(A + (size_t)(rowBase + r) * lda + kt + c);
            *(uint4*)(&As[buf][r * LSTR + c]) = x;
            __builtin_prefetch((const void*)(A + (size_t)(rowBase + r) * lda + kt + GBK + c), 0, 1);
        }
        #pragma unroll
        for (int j = 0; j < NB; ++j) {
            const int idx = tid + j * 256;
            const int r = idx >> 2, c = (idx & 3) * 8;
            uint4 x = *(const uint4*)(Bt + (size_t)(colBase + r) * ldb + kt + c);
            *(uint4*)(&Bs[buf][r * LSTR + c]) = x;
            __builtin_prefetch((const void*)(Bt + (size_t)(colBase + r) * ldb + kt + GBK + c), 0, 1);
        }
#endif
    };

    stage(0, 0);   // prologue: tile 0 -> buffer 0

    for (int it = 0; it < nt; ++it) {
        const int cur = it & 1;
#if USE_ASYNC_LDS
        __builtin_amdgcn_s_wait_asynccnt(0);   // my tile-`it` loads complete
#endif
        __syncthreads();                        // all waves' loads complete AND
                                                // all waves done with buf `cur^1`
        if (it + 1 < nt) stage(it + 1, cur ^ 1); // overlaps compute below

        // ---- fragments (documented wave32 layouts) ----
        const u16* Ab = As[cur];
        const u16* Bb = Bs[cur];
        v16bf afr[TM], bfr[TN];
        #pragma unroll
        for (int tm = 0; tm < TM; ++tm) {
            const int m  = wm * (TM * 16) + tm * 16 + lr16;
            const int ka = hiHalf ? 8 : 0;       // lanes16-31: K=8..15 / 24..31
            FragCast fc;
            fc.q[0] = *(const uint4*)(&Ab[m * LSTR + ka]);
            fc.q[1] = *(const uint4*)(&Ab[m * LSTR + ka + 16]);
            afr[tm] = fc.b;
        }
        #pragma unroll
        for (int tn = 0; tn < TN; ++tn) {
            const int n  = wn * (TN * 16) + tn * 16 + lr16;
            const int kb = hiHalf ? 8 : 0;       // lanes16-31: K=8..15 / 24..31
            FragCast fc;
            fc.q[0] = *(const uint4*)(&Bb[n * LSTR + kb]);
            fc.q[1] = *(const uint4*)(&Bb[n * LSTR + kb + 16]);
            bfr[tn] = fc.b;
        }
        #pragma unroll
        for (int tm = 0; tm < TM; ++tm)
            #pragma unroll
            for (int tn = 0; tn < TN; ++tn)
                acc[tm][tn] = __builtin_amdgcn_wmma_f32_16x16x32_bf16(
                    false, afr[tm], false, bfr[tn],
                    (short)0, acc[tm][tn], false, false);
    }

    // ---- epilogue: C layout VGPR r -> M=r(+8 for lanes>=16), lane&15 -> N ----
    const int lrow = hiHalf ? 8 : 0;
    #pragma unroll
    for (int tm = 0; tm < TM; ++tm) {
        #pragma unroll
        for (int tn = 0; tn < TN; ++tn) {
            const int gr = rowBase + wm * (TM * 16) + tm * 16 + lrow;
            const int gc = colBase + wn * (TN * 16) + tn * 16 + lr16;
            const float bia = bias ? bias[gc] : 0.0f;
            #pragma unroll
            for (int r = 0; r < 8; ++r) {
                float v = acc[tm][tn][r] + bia;
                if (relu) v = fmaxf(v, 0.0f);
                const int row = gr + r;
                if (resid) v += resid[(size_t)row * ldr + gc];
                if (outF)  outF[(size_t)row * ldc + gc] = v;
                else       outB[(size_t)row * ldc + gc] = f32_to_bf16(v);
            }
        }
    }
}

// ---------------------------------------------------------------------------
// Embedding + positional encoding: fp32 h and bf16 copy.
// ---------------------------------------------------------------------------
__global__ __launch_bounds__(256) void k_embed(
    const int* __restrict__ x, const float* __restrict__ emb,
    const float* __restrict__ pos, float* __restrict__ h, u16* __restrict__ hb)
{
    const int i = blockIdx.x;           // 0..4095
    const int tok = x[i];
    const int t = i % SEQ_T;
    for (int d = threadIdx.x; d < D_MODEL; d += 256) {
        float v = emb[(size_t)tok * D_MODEL + d] + pos[(size_t)t * D_MODEL + d];
        h [(size_t)i * D_MODEL + d] = v;
        hb[(size_t)i * D_MODEL + d] = f32_to_bf16(v);
    }
}

// W[K][N] fp32 -> Wt[N][K] bf16
__global__ __launch_bounds__(256) void k_transpose_w(
    const float* __restrict__ W, u16* __restrict__ Wt, int K, int N)
{
    const size_t idx = (size_t)blockIdx.x * 256 + threadIdx.x;
    if (idx >= (size_t)K * N) return;
    const int n = (int)(idx / K);
    const int k = (int)(idx % K);
    Wt[idx] = f32_to_bf16(W[(size_t)k * N + n]);
}

// fp32 -> bf16 elementwise (emb head matrix is already [N][K])
__global__ __launch_bounds__(256) void k_f32_to_bf16(
    const float* __restrict__ s, u16* __restrict__ d, size_t n)
{
    const size_t idx = (size_t)blockIdx.x * 256 + threadIdx.x;
    if (idx < n) d[idx] = f32_to_bf16(s[idx]);
}

// v bf16 [B*T, H*64] -> vt bf16 [B, H, 64, T]  (so P@V uses contiguous-K Bt)
__global__ __launch_bounds__(256) void k_transpose_v(
    const u16* __restrict__ v, u16* __restrict__ vt)
{
    const size_t total = (size_t)BATCH * NHEAD * DHEAD * SEQ_T;
    const size_t idx = (size_t)blockIdx.x * 256 + threadIdx.x;
    if (idx >= total) return;
    const int t  = (int)(idx & (SEQ_T - 1));
    size_t rest  = idx >> 11;
    const int d  = (int)(rest & (DHEAD - 1));
    rest >>= 6;
    const int hh = (int)(rest % NHEAD);
    const int b  = (int)(rest / NHEAD);
    vt[idx] = v[((size_t)b * SEQ_T + t) * D_MODEL + hh * DHEAD + d];
}

// ---------------------------------------------------------------------------
// Causal softmax over one (b,h): S fp32 [T,T] -> P bf16 [T,T], scale 1/8.
// ---------------------------------------------------------------------------
__global__ __launch_bounds__(256) void k_softmax_causal(
    const float* __restrict__ S, u16* __restrict__ P)
{
    const int t = blockIdx.x;
    const float* row = S + (size_t)t * SEQ_T;
    u16* prow = P + (size_t)t * SEQ_T;
    __shared__ float red[256];
    const float scale = 0.125f;   // 1/sqrt(64)

    float lmax = -3.4e38f;
    for (int c = threadIdx.x; c < SEQ_T; c += 256)
        if (c <= t) lmax = fmaxf(lmax, row[c] * scale);
    red[threadIdx.x] = lmax;
    __syncthreads();
    for (int s = 128; s > 0; s >>= 1) {
        if (threadIdx.x < s) red[threadIdx.x] = fmaxf(red[threadIdx.x], red[threadIdx.x + s]);
        __syncthreads();
    }
    const float m = red[0];
    __syncthreads();

    float lsum = 0.0f;
    for (int c = threadIdx.x; c < SEQ_T; c += 256)
        if (c <= t) lsum += __expf(row[c] * scale - m);
    red[threadIdx.x] = lsum;
    __syncthreads();
    for (int s = 128; s > 0; s >>= 1) {
        if (threadIdx.x < s) red[threadIdx.x] += red[threadIdx.x + s];
        __syncthreads();
    }
    const float inv = 1.0f / red[0];

    for (int c = threadIdx.x; c < SEQ_T; c += 256) {
        float p = (c <= t) ? __expf(row[c] * scale - m) * inv : 0.0f;
        prow[c] = f32_to_bf16(p);
    }
}

// ---------------------------------------------------------------------------
// LayerNorm over rows of 768; writes fp32 and bf16 copies.
// ---------------------------------------------------------------------------
__global__ __launch_bounds__(256) void k_layernorm(
    const float* __restrict__ in, const float* __restrict__ g,
    const float* __restrict__ b, float* __restrict__ outF, u16* __restrict__ outB)
{
    const int r = blockIdx.x;
    const float* row = in + (size_t)r * D_MODEL;
    __shared__ float red[256];

    float s = 0.0f;
    for (int d = threadIdx.x; d < D_MODEL; d += 256) s += row[d];
    red[threadIdx.x] = s;
    __syncthreads();
    for (int k = 128; k > 0; k >>= 1) {
        if (threadIdx.x < k) red[threadIdx.x] += red[threadIdx.x + k];
        __syncthreads();
    }
    const float mean = red[0] * (1.0f / D_MODEL);
    __syncthreads();

    float vs = 0.0f;
    for (int d = threadIdx.x; d < D_MODEL; d += 256) {
        float dv = row[d] - mean;
        vs += dv * dv;
    }
    red[threadIdx.x] = vs;
    __syncthreads();
    for (int k = 128; k > 0; k >>= 1) {
        if (threadIdx.x < k) red[threadIdx.x] += red[threadIdx.x + k];
        __syncthreads();
    }
    const float inv = rsqrtf(red[0] * (1.0f / D_MODEL) + 1e-5f);

    for (int d = threadIdx.x; d < D_MODEL; d += 256) {
        float o = (row[d] - mean) * inv * g[d] + b[d];
        outF[(size_t)r * D_MODEL + d] = o;
        outB[(size_t)r * D_MODEL + d] = f32_to_bf16(o);
    }
}

// ---------------------------------------------------------------------------
// Host orchestration
// ---------------------------------------------------------------------------
extern "C" void kernel_launch(void* const* d_in, const int* in_sizes, int n_in,
                              void* d_out, int out_size, void* d_ws, size_t ws_size,
                              hipStream_t stream)
{
    (void)in_sizes; (void)n_in; (void)out_size; (void)ws_size;

    const int*   x     = (const int*)  d_in[0];
    const float* emb   = (const float*)d_in[1];
    const float* pos   = (const float*)d_in[2];
    const float* Wq    = (const float*)d_in[3];
    const float* Wk    = (const float*)d_in[4];
    const float* Wv    = (const float*)d_in[5];
    const float* Wo    = (const float*)d_in[6];
    const float* ln1_g = (const float*)d_in[7];
    const float* ln1_b = (const float*)d_in[8];
    const float* ln2_g = (const float*)d_in[9];
    const float* ln2_b = (const float*)d_in[10];
    const float* ff1_w = (const float*)d_in[11];
    const float* ff1_b = (const float*)d_in[12];
    const float* ff2_w = (const float*)d_in[13];
    const float* ff2_b = (const float*)d_in[14];
    float* logits = (float*)d_out;

    // ---- workspace carve-up ----
    char* p = (char*)d_ws;
    auto take = [&](size_t bytes) { char* q = p; p += (bytes + 255) & ~(size_t)255; return q; };
    float* h      = (float*)take((size_t)MROWS * D_MODEL * 4);
    float* tmp    = (float*)take((size_t)MROWS * D_MODEL * 4);
    float* hln    = (float*)take((size_t)MROWS * D_MODEL * 4);
    u16*   xbf    = (u16*)  take((size_t)MROWS * D_MODEL * 2);
    u16*   qbf    = (u16*)  take((size_t)MROWS * D_MODEL * 2);
    u16*   kbf    = (u16*)  take((size_t)MROWS * D_MODEL * 2);
    u16*   vbf    = (u16*)  take((size_t)MROWS * D_MODEL * 2);
    u16*   vt     = (u16*)  take((size_t)BATCH * NHEAD * DHEAD * SEQ_T * 2);
    u16*   attnbf = (u16*)  take((size_t)MROWS * D_MODEL * 2);
    float* Sbuf   = (float*)take((size_t)SEQ_T * SEQ_T * 4);
    u16*   Pbf    = (u16*)  take((size_t)SEQ_T * SEQ_T * 2);
    u16*   ffabf  = (u16*)  take((size_t)MROWS * D_FF * 2);
    u16*   wtbf   = (u16*)  take((size_t)D_FF * D_MODEL * 2);
    u16*   embbf  = (u16*)  take((size_t)VOCAB * D_MODEL * 2);

    auto gemm = [&](const u16* A, int lda, const u16* Bt, int ldb,
                    float* oF, u16* oB, int ldc,
                    const float* bias, const float* resid, int ldr,
                    int M, int N, int K, int relu) {
        if ((N & 127) == 0) {
            dim3 grid(N / 128, M / 128);
            k_gemm<2, 4, 4, 2><<<grid, 256, 0, stream>>>(A, lda, Bt, ldb, oF, oB, ldc,
                                                         bias, resid, ldr, K, relu);
        } else {
            dim3 grid(N / 64, M / 128);
            k_gemm<4, 2, 2, 2><<<grid, 256, 0, stream>>>(A, lda, Bt, ldb, oF, oB, ldc,
                                                         bias, resid, ldr, K, relu);
        }
    };
    auto transpose = [&](const float* W, u16* Wt, int K, int N) {
        size_t total = (size_t)K * N;
        k_transpose_w<<<dim3((unsigned)((total + 255) / 256)), 256, 0, stream>>>(W, Wt, K, N);
    };

    // embedding head matrix (already [N=vocab][K=d_model]) -> bf16, once
    {
        size_t total = (size_t)VOCAB * D_MODEL;
        k_f32_to_bf16<<<dim3((unsigned)((total + 255) / 256)), 256, 0, stream>>>(emb, embbf, total);
    }
    // token + positional embedding
    k_embed<<<dim3(MROWS), 256, 0, stream>>>(x, emb, pos, h, xbf);

    for (int l = 0; l < NLAYER; ++l) {
        const size_t wOff  = (size_t)l * D_MODEL * D_MODEL;
        const size_t f1Off = (size_t)l * D_MODEL * D_FF;
        const size_t f2Off = (size_t)l * D_FF * D_MODEL;
        const size_t vOff  = (size_t)l * D_MODEL;

        // Q = x@Wq, K = x@Wk, V = x@Wv  (bf16 outputs, head-strided views later)
        transpose(Wq + wOff, wtbf, D_MODEL, D_MODEL);
        gemm(xbf, D_MODEL, wtbf, D_MODEL, nullptr, qbf, D_MODEL,
             nullptr, nullptr, 0, MROWS, D_MODEL, D_MODEL, 0);
        transpose(Wk + wOff, wtbf, D_MODEL, D_MODEL);
        gemm(xbf, D_MODEL, wtbf, D_MODEL, nullptr, kbf, D_MODEL,
             nullptr, nullptr, 0, MROWS, D_MODEL, D_MODEL, 0);
        transpose(Wv + wOff, wtbf, D_MODEL, D_MODEL);
        gemm(xbf, D_MODEL, wtbf, D_MODEL, nullptr, vbf, D_MODEL,
             nullptr, nullptr, 0, MROWS, D_MODEL, D_MODEL, 0);

        // V^T per head for the P@V GEMM
        {
            size_t total = (size_t)BATCH * NHEAD * DHEAD * SEQ_T;
            k_transpose_v<<<dim3((unsigned)((total + 255) / 256)), 256, 0, stream>>>(vbf, vt);
        }

        // attention per (b, head); S/P stay L2-resident (25 MB << 192 MB L2)
        for (int b = 0; b < BATCH; ++b) {
            for (int hh = 0; hh < NHEAD; ++hh) {
                const size_t hOff = (size_t)b * SEQ_T * D_MODEL + (size_t)hh * DHEAD;
                // S = Q_h @ K_h^T    (Bt := K_h directly, rows n with contiguous k)
                gemm(qbf + hOff, D_MODEL, kbf + hOff, D_MODEL,
                     Sbuf, nullptr, SEQ_T, nullptr, nullptr, 0,
                     SEQ_T, SEQ_T, DHEAD, 0);
                // P = softmax(S / 8, causal)
                k_softmax_causal<<<dim3(SEQ_T), 256, 0, stream>>>(Sbuf, Pbf);
                // A_h = P @ V_h      (Bt := V_h^T [64][T])
                gemm(Pbf, SEQ_T,
                     vt + ((size_t)b * NHEAD + hh) * DHEAD * SEQ_T, SEQ_T,
                     nullptr, attnbf + hOff, D_MODEL, nullptr, nullptr, 0,
                     SEQ_T, DHEAD, SEQ_T, 0);
            }
        }

        // O-projection with fused residual:  tmp = h + A @ Wo
        transpose(Wo + wOff, wtbf, D_MODEL, D_MODEL);
        gemm(attnbf, D_MODEL, wtbf, D_MODEL, tmp, nullptr, D_MODEL,
             nullptr, h, D_MODEL, MROWS, D_MODEL, D_MODEL, 0);

        // hln = LN1(tmp)
        k_layernorm<<<dim3(MROWS), 256, 0, stream>>>(tmp, ln1_g + vOff, ln1_b + vOff, hln, xbf);

        // FF1: relu(hln @ W1 + b1) -> bf16
        transpose(ff1_w + f1Off, wtbf, D_MODEL, D_FF);
        gemm(xbf, D_MODEL, wtbf, D_MODEL, nullptr, ffabf, D_FF,
             ff1_b + (size_t)l * D_FF, nullptr, 0, MROWS, D_FF, D_MODEL, 1);

        // FF2 with fused bias + residual: tmp = hln + ffa @ W2 + b2
        transpose(ff2_w + f2Off, wtbf, D_FF, D_MODEL);
        gemm(ffabf, D_FF, wtbf, D_FF, tmp, nullptr, D_MODEL,
             ff2_b + vOff, hln, D_MODEL, MROWS, D_MODEL, D_FF, 0);

        // h = LN2(tmp)  (fp32 for next residual, bf16 for next GEMMs)
        k_layernorm<<<dim3(MROWS), 256, 0, stream>>>(tmp, ln2_g + vOff, ln2_b + vOff, h, xbf);
    }

    // logits = h @ emb^T   (emb is already [vocab][d_model] = Bt layout)
    gemm(xbf, D_MODEL, embbf, D_MODEL, logits, nullptr, VOCAB,
         nullptr, nullptr, 0, MROWS, VOCAB, D_MODEL, 0);
}
